// MultiheadAttention_42606075576664
// MI455X (gfx1250) — matrix operand order repaired and checked by reading, not compiled, not run on hardware
//
#include <hip/hip_runtime.h>

#define NBATCH 2
#define SEQ    2048
#define EMBED  1024
#define NHEAD  16
#define HDIM   64
#define SROW   2065   // padded LDS row stride in floats (2065 mod 64 = 17, coprime -> conflict-free column reads)
#define AROW   1032   // padded LDS row stride in halves for fc A tile (2064 B)

typedef _Float16 v16h __attribute__((ext_vector_type(16)));
typedef _Float16 v8h  __attribute__((ext_vector_type(8)));
typedef float    v8f  __attribute__((ext_vector_type(8)));

union HVec {
    v16h v;
    v8h  h[2];
    _Float16 e[16];
};

// ---------------------------------------------------------------------------
// Stage 1: per-head projection  out[n,h,l,i] = sum_j X[n,l,h*64+j] * W[i,j] + b[i]
// Qp/Kp stored row-major [n,h,l,d] (f16); V stored transposed [n,h,d,l] (f16).
// ---------------------------------------------------------------------------
__global__ __launch_bounds__(256) void mha_proj(const float* __restrict__ X,
                                                const float* __restrict__ W,
                                                const float* __restrict__ bias,
                                                _Float16* __restrict__ Out,
                                                int transposeOut)
{
    __shared__ float WsT[HDIM * HDIM];   // transposed weights: WsT[j*64+i] = W[i*64+j]
    __shared__ float Xs[256];

    int tid = threadIdx.x;
    unsigned t = blockIdx.x * 256u + tid;
    int i = t & 63;
    int h = (t >> 6) & 15;
    int l = (t >> 10) & 2047;
    int n = (t >> 21) & 1;
    int h0 = (int)(((blockIdx.x * 256u) >> 6) & 15);   // first head covered by this block

    for (int kk = tid; kk < HDIM * HDIM; kk += 256)
        WsT[(kk & 63) * 64 + (kk >> 6)] = W[kk];
    Xs[tid] = X[(size_t)(n * SEQ + l) * EMBED + h0 * HDIM + tid];
    __syncthreads();

    int hL = h - h0;
    float acc = bias[i];
    #pragma unroll
    for (int j = 0; j < HDIM; ++j)
        acc += Xs[hL * HDIM + j] * WsT[j * 64 + i];

    size_t oidx = transposeOut
        ? ((size_t)((n * NHEAD + h) * HDIM + i) * SEQ + l)
        : ((size_t)((n * NHEAD + h) * SEQ + l) * HDIM + i);
    Out[oidx] = (_Float16)acc;
}

// ---------------------------------------------------------------------------
// Convert Wfc (fp32 row-major [i][j]) to f16; row-major Wfc is exactly the
// transposed-B layout for y = x @ Wfc^T.
// ---------------------------------------------------------------------------
__global__ __launch_bounds__(256) void mha_cvt(const float* __restrict__ src,
                                               _Float16* __restrict__ dst)
{
    unsigned t = blockIdx.x * 256u + threadIdx.x;
    dst[t] = (_Float16)src[t];
}

// ---------------------------------------------------------------------------
// Stage 2: attention. One block = (n, head, 16-row q-tile). 8 waves, each wave
// owns a 256-column slab of the 2048-wide score row block.
// ---------------------------------------------------------------------------
__global__ __launch_bounds__(256) void mha_attn(const _Float16* __restrict__ Qp,
                                                const _Float16* __restrict__ Kp,
                                                const _Float16* __restrict__ VpT,
                                                const int* __restrict__ mask,
                                                _Float16* __restrict__ Ctx)
{
    extern __shared__ char smem[];
    float* S     = (float*)smem;               // 16 x SROW scores
    float* Opart = S + 16 * SROW;              // 8 x 16 x 64 partial O per wave
    float* red   = Opart + 8 * 16 * 64;        // 16 x 17 reduction scratch
    float* rmax  = red + 16 * 17;              // 16
    float* rsum  = rmax + 16;                  // 16

    int tid  = threadIdx.x;
    int wave = tid >> 5;
    int lane = tid & 31;
    int hi   = lane >> 4;
    int ln   = lane & 15;

    int b  = blockIdx.x;
    int qt = b & 127;
    int hd = (b >> 7) & 15;
    int n  = (b >> 11) & 1;

    const float scale = 0.03125f;   // 1/sqrt(EMBED) = 1/32

    const _Float16* Qbase = Qp  + (size_t)(n * NHEAD + hd) * SEQ * HDIM;
    const _Float16* Kbase = Kp  + (size_t)(n * NHEAD + hd) * SEQ * HDIM;
    const _Float16* Vbase = VpT + (size_t)(n * NHEAD + hd) * HDIM * SEQ;

    // Q A-fragments for this q-tile (A 16x32 layout: lane M=ln, 16 contiguous K at hi*16)
    HVec aq0, aq1;
    {
        const _Float16* qrow = Qbase + (size_t)(qt * 16 + ln) * HDIM;
        aq0.h[0] = *(const v8h*)(qrow + hi * 16);
        aq0.h[1] = *(const v8h*)(qrow + hi * 16 + 8);
        aq1.h[0] = *(const v8h*)(qrow + 32 + hi * 16);
        aq1.h[1] = *(const v8h*)(qrow + 40 + hi * 16);
    }

    // ---- scores: S[q, k] = mask ? (q . k) * scale : -1e20 ----
    for (int ct = 0; ct < 16; ++ct) {
        int col0 = wave * 256 + ct * 16;
        // B 32x16 layout: lane N=ln (key position), 16 contiguous K(d) values at hi*16
        const _Float16* krow = Kbase + (size_t)(col0 + ln) * HDIM;
        HVec b0, b1;
        b0.v = *(const v16h*)(krow + hi * 16);          // d = 0..31
        b1.v = *(const v16h*)(krow + 32 + hi * 16);     // d = 32..63
        v8f c = {};
        c = __builtin_amdgcn_wmma_f32_16x16x32_f16(false, aq0.v, false, b0.v, (short)0, c, false, false);
        c = __builtin_amdgcn_wmma_f32_16x16x32_f16(false, aq1.v, false, b1.v, (short)0, c, false, false);

        int M0   = hi * 8;
        int ncol = col0 + ln;
        #pragma unroll
        for (int r = 0; r < 8; ++r) {
            int qrow = qt * 16 + M0 + r;
            int mv = mask[((size_t)n * SEQ + qrow) * SEQ + ncol];
            S[(M0 + r) * SROW + ncol] = mv ? c[r] * scale : -1.0e20f;
        }
    }
    __syncthreads();

    // ---- full-row softmax over 2048 columns; 16 threads per row, 128 cols each ----
    {
        int row = tid >> 4;
        int ch  = tid & 15;
        float* srow = S + row * SROW + ch * 128;

        float m = -3.0e38f;
        for (int i2 = 0; i2 < 128; ++i2) m = fmaxf(m, srow[i2]);
        red[row * 17 + ch] = m;
        __syncthreads();
        if (ch == 0) {
            float mm = red[row * 17];
            for (int j = 1; j < 16; ++j) mm = fmaxf(mm, red[row * 17 + j]);
            rmax[row] = mm;
        }
        __syncthreads();

        float mm = rmax[row];
        float sum = 0.0f;
        for (int i2 = 0; i2 < 128; ++i2) {
            float e2 = __expf(srow[i2] - mm);
            srow[i2] = e2;
            sum += e2;
        }
        red[row * 17 + ch] = sum;
        __syncthreads();
        if (ch == 0) {
            float ss = 0.0f;
            for (int j = 0; j < 16; ++j) ss += red[row * 17 + j];
            rsum[row] = ss;
        }
        __syncthreads();

        float inv = 1.0f / rsum[row];
        for (int i2 = 0; i2 < 128; ++i2) srow[i2] *= inv;
    }
    __syncthreads();

    // ---- attn @ V: each wave reduces its 256-col slab into partial O (16x64) ----
    v8f o0 = {}, o1 = {}, o2 = {}, o3 = {};
    for (int kt2 = 0; kt2 < 8; ++kt2) {
        int kcol = wave * 256 + kt2 * 32;
        // A-fragment of P from LDS (f32 -> f16): lane M=ln, 16 contiguous K at hi*16
        HVec ap;
        const float* sr = S + ln * SROW + kcol;
        #pragma unroll
        for (int j = 0; j < 8; ++j) {
            ap.e[j]     = (_Float16)sr[hi * 16 + j];
            ap.e[8 + j] = (_Float16)sr[hi * 16 + 8 + j];
        }
        // B-fragments of V from transposed layout: lane N=d, 16 contiguous kpos at hi*16
        const _Float16* vb = Vbase + kcol + hi * 16;
        HVec bv0, bv1, bv2, bv3;
        bv0.v = *(const v16h*)(vb + (size_t)(0 * 16 + ln) * SEQ);
        bv1.v = *(const v16h*)(vb + (size_t)(1 * 16 + ln) * SEQ);
        bv2.v = *(const v16h*)(vb + (size_t)(2 * 16 + ln) * SEQ);
        bv3.v = *(const v16h*)(vb + (size_t)(3 * 16 + ln) * SEQ);
        o0 = __builtin_amdgcn_wmma_f32_16x16x32_f16(false, ap.v, false, bv0.v, (short)0, o0, false, false);
        o1 = __builtin_amdgcn_wmma_f32_16x16x32_f16(false, ap.v, false, bv1.v, (short)0, o1, false, false);
        o2 = __builtin_amdgcn_wmma_f32_16x16x32_f16(false, ap.v, false, bv2.v, (short)0, o2, false, false);
        o3 = __builtin_amdgcn_wmma_f32_16x16x32_f16(false, ap.v, false, bv3.v, (short)0, o3, false, false);
    }
    __syncthreads();   // all waves done reading S

    {
        int M0 = hi * 8;
        #pragma unroll
        for (int r = 0; r < 8; ++r) {
            Opart[wave * 1024 + (M0 + r) * 64 +  0 + ln] = o0[r];
            Opart[wave * 1024 + (M0 + r) * 64 + 16 + ln] = o1[r];
            Opart[wave * 1024 + (M0 + r) * 64 + 32 + ln] = o2[r];
            Opart[wave * 1024 + (M0 + r) * 64 + 48 + ln] = o3[r];
        }
    }
    __syncthreads();

    // ---- reduce 8 wave partials, write context (f16, [n, l, h*64+d]) ----
    #pragma unroll
    for (int e2 = 0; e2 < 4; ++e2) {
        int idx = tid * 4 + e2;
        int row = idx >> 6;
        int d   = idx & 63;
        float s = 0.0f;
        #pragma unroll
        for (int w2 = 0; w2 < 8; ++w2) s += Opart[w2 * 1024 + idx];
        Ctx[(size_t)(n * SEQ + qt * 16 + row) * EMBED + hd * HDIM + d] = (_Float16)s;
    }
}

// ---------------------------------------------------------------------------
// Stage 3: out = Ctx @ Wfc^T + bfc  (4096x1024 @ 1024x1024), fp32 output.
// The 16x1024 A tile is shared by all 8 waves of the block -> stage it once
// into LDS with the CDNA5 async global->LDS path (ASYNCcnt), then feed
// A-fragments from LDS while B streams from global.
// ---------------------------------------------------------------------------
__global__ __launch_bounds__(256) void mha_fc(const _Float16* __restrict__ Ctx,
                                              const _Float16* __restrict__ W16,
                                              const float* __restrict__ bfc,
                                              float* __restrict__ out)
{
    __shared__ _Float16 As[16 * AROW];   // 16 rows, padded stride 1032 halves (2064 B)

    int tid  = threadIdx.x;
    int wave = tid >> 5;
    int lane = tid & 31;
    int hi   = lane >> 4;
    int ln   = lane & 15;

    int rt = blockIdx.x;                 // row tile (16 rows of N*L=4096)
    int ct = blockIdx.y * 8 + wave;      // col tile (16 cols of 1024)

    // ---- async-stage the 16x1024 f16 A tile into LDS (32 KB, 2048 x b128) ----
    typedef __attribute__((address_space(3))) _Float16 lds_f16;
    lds_f16* AsL = (lds_f16*)As;         // raw LDS offset space
    #pragma unroll
    for (int c0 = 0; c0 < 8; ++c0) {
        int c    = c0 * 256 + tid;
        int row  = c >> 7;
        int col8 = c & 127;
        const _Float16* g = Ctx + (size_t)(rt * 16 + row) * EMBED + col8 * 8;
        unsigned loff = (unsigned)(size_t)(AsL + row * AROW + col8 * 8);
        asm volatile("global_load_async_to_lds_b128 %0, %1, off"
                     :: "v"(loff), "v"(g) : "memory");
    }
    asm volatile("s_wait_asynccnt 0x0" ::: "memory");
    __syncthreads();

    const _Float16* arow = As + (size_t)ln * AROW;
    const _Float16* brow = W16 + (size_t)(ct * 16 + ln) * EMBED;

    v8f c = {};
    for (int kt = 0; kt < 32; ++kt) {
        HVec a, bb;
        a.h[0] = *(const v8h*)(arow + kt * 32 + hi * 16);       // ds_load_b128
        a.h[1] = *(const v8h*)(arow + kt * 32 + hi * 16 + 8);
        bb.v   = *(const v16h*)(brow + kt * 32 + hi * 16);      // global b128 x2
        c = __builtin_amdgcn_wmma_f32_16x16x32_f16(false, a.v, false, bb.v, (short)0, c, false, false);
    }

    float bias = bfc[ct * 16 + ln];
    #pragma unroll
    for (int r = 0; r < 8; ++r)
        out[(size_t)(rt * 16 + hi * 8 + r) * EMBED + ct * 16 + ln] = c[r] + bias;
}

// ---------------------------------------------------------------------------
extern "C" void kernel_launch(void* const* d_in, const int* in_sizes, int n_in,
                              void* d_out, int out_size, void* d_ws, size_t ws_size,
                              hipStream_t stream) {
    const float* q    = (const float*)d_in[0];
    const float* k    = (const float*)d_in[1];
    const float* v    = (const float*)d_in[2];
    const int*   mask = (const int*)d_in[3];
    const float* Wq   = (const float*)d_in[4];
    const float* bq   = (const float*)d_in[5];
    const float* Wk   = (const float*)d_in[6];
    const float* bk   = (const float*)d_in[7];
    const float* Wv   = (const float*)d_in[8];
    const float* bv   = (const float*)d_in[9];
    const float* Wfc  = (const float*)d_in[10];
    const float* bfc  = (const float*)d_in[11];
    float* out = (float*)d_out;

    char* ws = (char*)d_ws;
    _Float16* Qp  = (_Float16*)(ws);                          //  8 MB
    _Float16* Kp  = (_Float16*)(ws + ((size_t)8  << 20));     //  8 MB
    _Float16* VpT = (_Float16*)(ws + ((size_t)16 << 20));     //  8 MB
    _Float16* Ctx = (_Float16*)(ws + ((size_t)24 << 20));     //  8 MB
    _Float16* W16 = (_Float16*)(ws + ((size_t)32 << 20));     //  2 MB

    // Stage 1: projections (N*L*H*D = 4,194,304 threads per matrix)
    mha_proj<<<16384, 256, 0, stream>>>(q, Wq, bq, Qp, 0);
    mha_proj<<<16384, 256, 0, stream>>>(k, Wk, bk, Kp, 0);
    mha_proj<<<16384, 256, 0, stream>>>(v, Wv, bv, VpT, 1);
    mha_cvt<<<4096, 256, 0, stream>>>(Wfc, W16);

    // Stage 2: attention. One block per (n, head, q-tile): 2*16*128 = 4096 blocks.
    size_t smemBytes = (size_t)(16 * SROW + 8 * 16 * 64 + 16 * 17 + 32) * sizeof(float);
    mha_attn<<<4096, 256, smemBytes, stream>>>(Qp, Kp, VpT, mask, Ctx);

    // Stage 3: FC. 256 row tiles x (8 blocks * 8 waves = 64 col tiles).
    mha_fc<<<dim3(256, 8), 256, 0, stream>>>(Ctx, W16, bfc, out);
}
